// MHA_77472620085876
// MI455X (gfx1250) — matrix operand order repaired and checked by reading, not compiled, not run on hardware
//
#include <hip/hip_runtime.h>

// ---------------------------------------------------------------------------
// Causal MHA forward for gfx1250 (MI455X): bf16 WMMA everywhere, fp32 accum.
// B=4, L=2048, D=1024, H=16, hd=64.
// Double-buffered LDS staging, async global->LDS copies when available.
// ---------------------------------------------------------------------------

typedef __bf16 bf16;
typedef __attribute__((ext_vector_type(16))) __bf16 bf16x16;
typedef __attribute__((ext_vector_type(8)))  __bf16 bf16x8;
typedef __attribute__((ext_vector_type(8)))  float  f32x8;
typedef __attribute__((ext_vector_type(4)))  int    i32x4;

#define MHA_B   4
#define MHA_L   2048
#define MHA_D   1024
#define MHA_H   16
#define MHA_HD  64
#define MHA_M   (MHA_B * MHA_L)   // 8192
#define MHA_3D  (3 * MHA_D)       // 3072

// ---------------------- async global->LDS copy (guarded) -------------------

#if __has_builtin(__builtin_amdgcn_global_load_async_to_lds_b128) && \
    __has_builtin(__builtin_amdgcn_s_wait_asynccnt)
#define USE_ASYNC_LDS 1
#else
#define USE_ASYNC_LDS 0
#endif

// Builtin signature (from hipcc diagnostic): (int4 addrspace(1)*, int4
// addrspace(3)*, imm offset, imm cpol).
typedef __attribute__((address_space(1))) i32x4 gi32x4;
typedef __attribute__((address_space(3))) i32x4 si32x4;

// Copy 16 bytes (one b128 per lane) from global to LDS.
__device__ __forceinline__ void cp_b128(void* dst_lds, const void* src_gbl) {
#if USE_ASYNC_LDS
  __builtin_amdgcn_global_load_async_to_lds_b128((gi32x4*)src_gbl,
                                                 (si32x4*)dst_lds,
                                                 /*offset=*/0, /*cpol=*/0);
#else
  *(bf16x8*)dst_lds = *(const bf16x8*)src_gbl;
#endif
}

// Wait for all outstanding async copies from this wave.
__device__ __forceinline__ void cp_commit() {
#if USE_ASYNC_LDS
  __builtin_amdgcn_s_wait_asynccnt(0);
#endif
}

// ------------------------------ WMMA helpers -------------------------------

__device__ __forceinline__ f32x8 wmma_bf16(bf16x16 a, bf16x16 b, f32x8 c) {
  return __builtin_amdgcn_wmma_f32_16x16x32_bf16(
      /*neg_a=*/false, a, /*neg_b=*/false, b,
      /*c_mod=*/(short)0, c, /*reuse_a=*/false, /*reuse_b=*/false);
}

// A-matrix fragment (16x32 bf16): lane m = lane&15, half g = lane>>4.
// Lane holds the contiguous K run [16g, 16g+16); caller passes &row[16g].
__device__ __forceinline__ bf16x16 frag_a(const bf16* p) {
  bf16x8 lo = *(const bf16x8*)p;
  bf16x8 hi = *(const bf16x8*)(p + 8);
  bf16x16 f;
#pragma unroll
  for (int i = 0; i < 8; ++i) { f[i] = lo[i]; f[8 + i] = hi[i]; }
  return f;
}

// B-matrix fragment (32x16 bf16): lane n = lane&15, half g = lane>>4.
// Lane holds the contiguous K run [16g, 16g+16); caller passes &col[16g].
__device__ __forceinline__ bf16x16 frag_b(const bf16* p) {
  bf16x8 lo = *(const bf16x8*)p;
  bf16x8 hi = *(const bf16x8*)(p + 8);
  bf16x16 f;
#pragma unroll
  for (int i = 0; i < 8; ++i) { f[i] = lo[i]; f[8 + i] = hi[i]; }
  return f;
}

// ------------------------------ fp32 -> bf16 -------------------------------

__global__ __launch_bounds__(256)
void cvt_f32_bf16(const float* __restrict__ src, bf16* __restrict__ dst, long n) {
  long i = ((long)blockIdx.x * blockDim.x + threadIdx.x) * 4;
  if (i + 3 < n) {
    float4 f = *(const float4*)(src + i);
    dst[i + 0] = (bf16)f.x;
    dst[i + 1] = (bf16)f.y;
    dst[i + 2] = (bf16)f.z;
    dst[i + 3] = (bf16)f.w;
  } else {
    for (; i < n; ++i) dst[i] = (bf16)src[i];
  }
}

// ------------------------------ QKV projection -----------------------------
// qkv = x @ Wqkv^T + b.  A = x [M,K], B = Wqkv [N,K]; both fragments read
// contiguous K runs.  Epilogue scatters into q/k/v [B,H,L,hd] bf16 with the
// softmax scale folded into q.

#define GM_BM 128
#define GM_BN 64
#define GM_BK 32
#define GM_LDA 40   // bf16 elems per LDS row (32 + 8 pad, 80B: 16B multiple)

__global__ __launch_bounds__(256)
void qkv_gemm(const bf16* __restrict__ X,    // [8192, 1024]
              const bf16* __restrict__ W,    // [3072, 1024]
              const float* __restrict__ bias,// [3072]
              bf16* __restrict__ Qo, bf16* __restrict__ Ko, bf16* __restrict__ Vo) {
  __shared__ bf16 As[2][GM_BM][GM_LDA];   // 2 x 10240 B
  __shared__ bf16 Bs[2][GM_BN][GM_LDA];   // 2 x  5120 B

  const int K  = MHA_D;
  const int m0 = blockIdx.x * GM_BM;
  const int n0 = blockIdx.y * GM_BN;
  const int t    = threadIdx.x;
  const int wave = t >> 5;
  const int lane = t & 31;
  const int wm = (wave & 3) * 32;
  const int wn = (wave >> 2) * 32;
  const int lm = lane & 15;
  const int g  = lane >> 4;

  f32x8 acc[2][2];
#pragma unroll
  for (int i = 0; i < 2; ++i)
#pragma unroll
    for (int j = 0; j < 2; ++j) acc[i][j] = (f32x8){0, 0, 0, 0, 0, 0, 0, 0};

  const int arow = t >> 1, ahalf = (t & 1) * 16;     // 128x32, 16 bf16/thr
  const int brow = t >> 2, bq    = (t & 3) * 8;      // 64x32,   8 bf16/thr
  const bf16* aG = X + (long)(m0 + arow) * K + ahalf;
  const bf16* bG = W + (long)(n0 + brow) * K + bq;

  auto stage = [&](int buf, int k0) {
    cp_b128(&As[buf][arow][ahalf],     aG + k0);
    cp_b128(&As[buf][arow][ahalf + 8], aG + k0 + 8);
    cp_b128(&Bs[buf][brow][bq],        bG + k0);
  };

  stage(0, 0);
  cp_commit();
  __syncthreads();

  for (int k0 = 0; k0 < K; k0 += GM_BK) {
    const int cur = (k0 >> 5) & 1;
    const int nxt = cur ^ 1;
    if (k0 + GM_BK < K) stage(nxt, k0 + GM_BK);   // overlap with compute

    bf16x16 af[2], bfr[2];
#pragma unroll
    for (int tm = 0; tm < 2; ++tm)
      af[tm] = frag_a(&As[cur][wm + tm * 16 + lm][16 * g]);
#pragma unroll
    for (int tn = 0; tn < 2; ++tn)
      bfr[tn] = frag_b(&Bs[cur][wn + tn * 16 + lm][16 * g]);
#pragma unroll
    for (int tm = 0; tm < 2; ++tm)
#pragma unroll
      for (int tn = 0; tn < 2; ++tn)
        acc[tm][tn] = wmma_bf16(af[tm], bfr[tn], acc[tm][tn]);

    cp_commit();
    __syncthreads();
  }

  // Epilogue: bias + scatter into [B,H,L,hd]; scale q by 1/sqrt(64).
#pragma unroll
  for (int tm = 0; tm < 2; ++tm) {
#pragma unroll
    for (int tn = 0; tn < 2; ++tn) {
#pragma unroll
      for (int r = 0; r < 8; ++r) {
        int mg = m0 + wm + tm * 16 + r + 8 * g;
        int ng = n0 + wn + tn * 16 + lm;
        float val = acc[tm][tn][r] + bias[ng];
        int sel  = ng >> 10;          // 0=q 1=k 2=v
        int d    = ng & (MHA_D - 1);
        int h    = d >> 6;
        int hdix = d & (MHA_HD - 1);
        int bb   = mg >> 11;
        int ll   = mg & (MHA_L - 1);
        long idx = (((long)(bb * MHA_H + h)) * MHA_L + ll) * MHA_HD + hdix;
        if (sel == 0)       Qo[idx] = (bf16)(val * 0.125f);
        else if (sel == 1)  Ko[idx] = (bf16)val;
        else                Vo[idx] = (bf16)val;
      }
    }
  }
}

// ------------------------------ flash attention ----------------------------
// Block: one (b,h), 128 queries; 8 waves x 16 queries. Double-buffered 32-key
// K/V tiles in LDS (V transposed), online softmax, all matmuls via bf16 WMMA.

__global__ __launch_bounds__(256)
void attn_fwd(const bf16* __restrict__ Q,   // [B*H, L, 64] (pre-scaled)
              const bf16* __restrict__ Kd,  // [B*H, L, 64]
              const bf16* __restrict__ V,   // [B*H, L, 64]
              bf16* __restrict__ AO) {      // [B, L, D]
  __shared__ bf16 Ks[2][32][72];    // [key][feat] pad -> 144B rows
  __shared__ bf16 Vt[2][64][40];    // [feat][key] transposed, pad -> 80B rows
  __shared__ bf16 Ps[8][16][40];    // per-wave P tile [qrow][key]

  const int bh = blockIdx.y;
  const int b  = bh >> 4;
  const int h  = bh & 15;
  const int q0 = blockIdx.x * 128;
  const int t    = threadIdx.x;
  const int wave = t >> 5;
  const int lane = t & 31;
  const int lm = lane & 15;
  const int g  = lane >> 4;
  const int qw = q0 + wave * 16;    // first query row of this wave

  const bf16* qrow = Q + ((long)bh * MHA_L + qw + lm) * MHA_HD;
  bf16x16 aq[2];
#pragma unroll
  for (int kk = 0; kk < 2; ++kk)
    aq[kk] = frag_a(qrow + kk * 32 + 16 * g);

  f32x8 o[4];
#pragma unroll
  for (int i = 0; i < 4; ++i) o[i] = (f32x8){0, 0, 0, 0, 0, 0, 0, 0};
  float mrow[8], lrow[8];
#pragma unroll
  for (int r = 0; r < 8; ++r) { mrow[r] = -3.0e38f; lrow[r] = 0.0f; }

  const bf16* kBase = Kd + (long)bh * MHA_L * MHA_HD;
  const bf16* vBase = V  + (long)bh * MHA_L * MHA_HD;
  const int krow = t >> 3;          // 0..31
  const int kseg = (t & 7) * 8;     // 0..56

  auto stage_kv = [&](int buf, int kb) {
    cp_b128(&Ks[buf][krow][kseg], kBase + (long)(kb + krow) * MHA_HD + kseg);
    bf16x8 vv = *(const bf16x8*)(vBase + (long)(kb + krow) * MHA_HD + kseg);
#pragma unroll
    for (int j = 0; j < 8; ++j) Vt[buf][kseg + j][krow] = vv[j];
  };

  const int kend = q0 + 128;        // causal limit for this block
  stage_kv(0, 0);
  cp_commit();
  __syncthreads();

  for (int kb = 0; kb < kend; kb += 32) {
    const int cur = (kb >> 5) & 1;
    const int nxt = cur ^ 1;
    if (kb + 32 < kend) stage_kv(nxt, kb + 32);

    if (kb <= qw + 15) {            // wave-uniform causal skip
      // S = q @ k^T for 32 keys: two 16x16 tiles, K=64 -> 2 WMMA each.
      f32x8 sc[2];
#pragma unroll
      for (int s = 0; s < 2; ++s) {
        sc[s] = (f32x8){0, 0, 0, 0, 0, 0, 0, 0};
#pragma unroll
        for (int kk = 0; kk < 2; ++kk) {
          bf16x16 bk = frag_b(&Ks[cur][s * 16 + lm][kk * 32 + 16 * g]);
          sc[s] = wmma_bf16(aq[kk], bk, sc[s]);
        }
      }

      // Causal mask + online softmax (rows live in 16-lane halves).
      float rs[8];
#pragma unroll
      for (int r = 0; r < 8; ++r) {
        int qg = qw + r + 8 * g;
        float s0 = sc[0][r], s1 = sc[1][r];
        if (kb + lm      > qg) s0 = -3.0e38f;
        if (kb + 16 + lm > qg) s1 = -3.0e38f;
        float vmax = fmaxf(s0, s1);
#pragma unroll
        for (int off = 1; off < 16; off <<= 1)
          vmax = fmaxf(vmax, __shfl_xor(vmax, off, 32));
        float mnew  = fmaxf(mrow[r], vmax);
        float scale = __expf(mrow[r] - mnew);
        mrow[r] = mnew;
        float p0 = __expf(s0 - mnew);
        float p1 = __expf(s1 - mnew);
        float ps = p0 + p1;
#pragma unroll
        for (int off = 1; off < 16; off <<= 1)
          ps += __shfl_xor(ps, off, 32);
        lrow[r] = lrow[r] * scale + ps;
        rs[r] = scale;
        Ps[wave][r + 8 * g][lm]      = (bf16)p0;
        Ps[wave][r + 8 * g][16 + lm] = (bf16)p1;
      }

      // Rescale O, then O += P @ V (K=32 keys, 4 hd tiles of 16).
#pragma unroll
      for (int tn = 0; tn < 4; ++tn)
#pragma unroll
        for (int r = 0; r < 8; ++r) o[tn][r] *= rs[r];

      bf16x16 pa = frag_a(&Ps[wave][lm][16 * g]);   // wave-local LDS, DS in-order
#pragma unroll
      for (int tn = 0; tn < 4; ++tn) {
        bf16x16 vb = frag_b(&Vt[cur][tn * 16 + lm][16 * g]);
        o[tn] = wmma_bf16(pa, vb, o[tn]);
      }
    }

    cp_commit();
    __syncthreads();
  }

  // Epilogue: O / l, write bf16 attn_out in [B, L, D] (head-concat) layout.
  bf16* aob = AO + (long)b * MHA_L * MHA_D + h * MHA_HD;
#pragma unroll
  for (int r = 0; r < 8; ++r) {
    float inv = 1.0f / lrow[r];
    int qg = qw + r + 8 * g;
#pragma unroll
    for (int tn = 0; tn < 4; ++tn)
      aob[(long)qg * MHA_D + tn * 16 + lm] = (bf16)(o[tn][r] * inv);
  }
}

// ------------------------------ output projection --------------------------

__global__ __launch_bounds__(256)
void out_gemm(const bf16* __restrict__ X,    // [8192, 1024] attn_out bf16
              const bf16* __restrict__ W,    // [1024, 1024] out_w
              const float* __restrict__ bias,// [1024]
              float* __restrict__ Out) {     // [8192, 1024] fp32
  __shared__ bf16 As[2][GM_BM][GM_LDA];
  __shared__ bf16 Bs[2][GM_BN][GM_LDA];

  const int K  = MHA_D;
  const int m0 = blockIdx.x * GM_BM;
  const int n0 = blockIdx.y * GM_BN;
  const int t    = threadIdx.x;
  const int wave = t >> 5;
  const int lane = t & 31;
  const int wm = (wave & 3) * 32;
  const int wn = (wave >> 2) * 32;
  const int lm = lane & 15;
  const int g  = lane >> 4;

  f32x8 acc[2][2];
#pragma unroll
  for (int i = 0; i < 2; ++i)
#pragma unroll
    for (int j = 0; j < 2; ++j) acc[i][j] = (f32x8){0, 0, 0, 0, 0, 0, 0, 0};

  const int arow = t >> 1, ahalf = (t & 1) * 16;
  const int brow = t >> 2, bq    = (t & 3) * 8;
  const bf16* aG = X + (long)(m0 + arow) * K + ahalf;
  const bf16* bG = W + (long)(n0 + brow) * K + bq;

  auto stage = [&](int buf, int k0) {
    cp_b128(&As[buf][arow][ahalf],     aG + k0);
    cp_b128(&As[buf][arow][ahalf + 8], aG + k0 + 8);
    cp_b128(&Bs[buf][brow][bq],        bG + k0);
  };

  stage(0, 0);
  cp_commit();
  __syncthreads();

  for (int k0 = 0; k0 < K; k0 += GM_BK) {
    const int cur = (k0 >> 5) & 1;
    const int nxt = cur ^ 1;
    if (k0 + GM_BK < K) stage(nxt, k0 + GM_BK);

    bf16x16 af[2], bfr[2];
#pragma unroll
    for (int tm = 0; tm < 2; ++tm)
      af[tm] = frag_a(&As[cur][wm + tm * 16 + lm][16 * g]);
#pragma unroll
    for (int tn = 0; tn < 2; ++tn)
      bfr[tn] = frag_b(&Bs[cur][wn + tn * 16 + lm][16 * g]);
#pragma unroll
    for (int tm = 0; tm < 2; ++tm)
#pragma unroll
      for (int tn = 0; tn < 2; ++tn)
        acc[tm][tn] = wmma_bf16(af[tm], bfr[tn], acc[tm][tn]);

    cp_commit();
    __syncthreads();
  }

#pragma unroll
  for (int tm = 0; tm < 2; ++tm) {
#pragma unroll
    for (int tn = 0; tn < 2; ++tn) {
#pragma unroll
      for (int r = 0; r < 8; ++r) {
        int mg = m0 + wm + tm * 16 + r + 8 * g;
        int ng = n0 + wn + tn * 16 + lm;
        Out[(long)mg * MHA_D + ng] = acc[tm][tn][r] + bias[ng];
      }
    }
  }
}

// ------------------------------ launcher -----------------------------------

extern "C" void kernel_launch(void* const* d_in, const int* in_sizes, int n_in,
                              void* d_out, int out_size, void* d_ws, size_t ws_size,
                              hipStream_t stream) {
  const float* x      = (const float*)d_in[0];
  const float* Wqkv_w = (const float*)d_in[1];
  const float* Wqkv_b = (const float*)d_in[2];
  const float* out_w  = (const float*)d_in[3];
  const float* out_b  = (const float*)d_in[4];
  float* out = (float*)d_out;

  const long NX  = (long)MHA_M * MHA_D;     // 8388608
  const long NWQ = (long)MHA_3D * MHA_D;    // 3145728
  const long NWO = (long)MHA_D * MHA_D;     // 1048576

  bf16* xb  = (bf16*)d_ws;
  bf16* wq  = xb  + NX;
  bf16* wo  = wq  + NWQ;
  bf16* Qb  = wo  + NWO;
  bf16* Kb  = Qb  + NX;
  bf16* Vb  = Kb  + NX;
  bf16* AOb = Vb  + NX;
  // total: 5*NX + NWQ + NWO elems = ~92 MB of workspace

  cvt_f32_bf16<<<(unsigned)((NX  + 1023) / 1024), 256, 0, stream>>>(x,      xb, NX);
  cvt_f32_bf16<<<(unsigned)((NWQ + 1023) / 1024), 256, 0, stream>>>(Wqkv_w, wq, NWQ);
  cvt_f32_bf16<<<(unsigned)((NWO + 1023) / 1024), 256, 0, stream>>>(out_w,  wo, NWO);

  qkv_gemm<<<dim3(MHA_M / GM_BM, MHA_3D / GM_BN), 256, 0, stream>>>(
      xb, wq, Wqkv_b, Qb, Kb, Vb);

  attn_fwd<<<dim3(MHA_L / 128, MHA_B * MHA_H), 256, 0, stream>>>(Qb, Kb, Vb, AOb);

  out_gemm<<<dim3(MHA_M / GM_BM, MHA_D / GM_BN), 256, 0, stream>>>(
      AOb, wo, out_b, out);
}
